// ScaledDotProductAttention_20126216749083
// MI455X (gfx1250) — compile-verified, matrix operand order checked by reading, not run on hardware
//
#include <hip/hip_runtime.h>

// Causal flash attention fwd: B=4, H=16, S=2048, D=128, fp32 in/out.
// Pre-pass converts K -> bf16 row-major and V -> bf16 transposed into d_ws,
// so the hot loop is pure async-DMA staging + ds_load_b128 + WMMA.
#define B_  4
#define H_  16
#define S_  2048
#define D_  128
#define BQ  128              // q rows per block (8 waves x 16 rows)
#define BC  32               // kv positions per pipeline stage
#define NW  8                // waves per block
#define KS2 136              // bf16 stride of K rows in LDS (128 + 8 pad)
#define VS2 40               // bf16 stride of V^T / P rows in LDS (32 + 8 pad)

typedef __bf16 bf16;
typedef __attribute__((ext_vector_type(16))) __bf16 v16bf;
typedef __attribute__((ext_vector_type(8)))  __bf16 v8bf;
typedef __attribute__((ext_vector_type(4)))  __bf16 v4bf;
typedef __attribute__((ext_vector_type(8)))  float  v8f;
typedef __attribute__((ext_vector_type(4)))  float  v4f;

static __device__ __forceinline__ v16bf cat8(v8bf lo, v8bf hi) {
  return __builtin_shufflevector(lo, hi, 0,1,2,3,4,5,6,7,8,9,10,11,12,13,14,15);
}

// ---------------- pre-pass 1: K fp32 -> bf16, same layout ----------------
__global__ __launch_bounds__(256)
void cvt_k_kernel(const float* __restrict__ kin, bf16* __restrict__ kout) {
  const size_t i = ((size_t)blockIdx.x * 256 + threadIdx.x) * 4;
  v4f x = *(const v4f*)(kin + i);
  v4bf y;
  #pragma unroll
  for (int j = 0; j < 4; ++j) y[j] = (bf16)x[j];
  *(v4bf*)(kout + i) = y;
}

// ------- pre-pass 2: V fp32 [bh][s][d] -> bf16 transposed [bh][d][s] -------
__global__ __launch_bounds__(256)
void cvt_vt_kernel(const float* __restrict__ vin, bf16* __restrict__ vtout) {
  __shared__ __align__(16) bf16 t[32 * KS2];   // [s][d] tile, padded
  const int bid   = blockIdx.x;                // bh * (S/32) + stile
  const int stile = bid & (S_ / 32 - 1);
  const int bh    = bid >> 6;                  // S/32 = 64
  const float* vp = vin + ((size_t)bh * S_ + (size_t)stile * 32) * D_;
  bf16* op = vtout + (size_t)bh * D_ * S_ + (size_t)stile * 32;
  const int tid = threadIdx.x;

  #pragma unroll
  for (int it = 0; it < 4; ++it) {             // 32x128 floats, float4 each
    const int i = it * 256 + tid;
    const int s = i >> 5;
    const int d = (i & 31) * 4;
    v4f x = *(const v4f*)(vp + (size_t)s * D_ + d);
    v4bf y;
    #pragma unroll
    for (int j = 0; j < 4; ++j) y[j] = (bf16)x[j];
    *(v4bf*)(&t[s * KS2 + d]) = y;
  }
  __syncthreads();
  #pragma unroll
  for (int it = 0; it < 4; ++it) {             // write 128 rows x 32 bf16
    const int i  = it * 256 + tid;             // v4bf groups: 128 d x 8
    const int d  = i >> 3;
    const int s4 = (i & 7) * 4;
    v4bf y;
    #pragma unroll
    for (int j = 0; j < 4; ++j) y[j] = t[(s4 + j) * KS2 + d];
    *(v4bf*)(op + (size_t)d * S_ + s4) = y;
  }
}

// ---------------------------- attention kernel ----------------------------
__global__ __launch_bounds__(256, 1)
void fa_fwd_kernel(const float* __restrict__ qg, const bf16* __restrict__ kws,
                   const bf16* __restrict__ vtws, float* __restrict__ og) {
  // Double-buffered bf16 stages, filled by global_load_async_to_lds_b128.
  __shared__ __align__(16) bf16 k_s[2][BC * KS2];   // K stage, row-major
  __shared__ __align__(16) bf16 vt_s[2][D_ * VS2];  // V stage, transposed
  __shared__ __align__(16) bf16 p_s[NW * 16 * VS2]; // per-wave P re-layout

  const int tid   = threadIdx.x;
  const int wid   = tid >> 5;
  const int lane  = tid & 31;
  const int lhalf = lane >> 4;      // WMMA K-group striping (lanes 0-15 vs 16-31)
  const int l16   = lane & 15;

  const int nqb  = S_ / BQ;
  const int qblk = blockIdx.x % nqb;
  const int bh   = blockIdx.x / nqb;
  const float* qp = qg + (size_t)bh * S_ * D_;
  const bf16*  kp = kws + (size_t)bh * S_ * D_;
  const bf16*  vp = vtws + (size_t)bh * D_ * S_;   // [d][s]
  float*       op = og + (size_t)bh * S_ * D_;

  const int qmin = qblk * BQ + wid * 16;
  const int qmax = qmin + 15;

  // fold 1/sqrt(D) and log2(e) into Q so softmax uses v_exp_f32 directly
  const float scale = 0.08838834764831845f * 1.4426950408889634f;

  // ---- Q A-fragments: four 16x32 bf16 chunks covering D=128 (one-time cvt) ----
  v16bf qf[4];
  {
    const float* qr = qp + (size_t)(qmin + l16) * D_;
    #pragma unroll
    for (int c = 0; c < 4; ++c) {
      const int d0 = 32 * c + lhalf * 8;
      v4f a0 = *(const v4f*)(qr + d0);
      v4f a1 = *(const v4f*)(qr + d0 + 4);
      v4f b0 = *(const v4f*)(qr + d0 + 16);
      v4f b1 = *(const v4f*)(qr + d0 + 20);
      v16bf f;
      #pragma unroll
      for (int j = 0; j < 4; ++j) {
        f[j]      = (bf16)(a0[j] * scale);
        f[4 + j]  = (bf16)(a1[j] * scale);
        f[8 + j]  = (bf16)(b0[j] * scale);
        f[12 + j] = (bf16)(b1[j] * scale);
      }
      qf[c] = f;
    }
  }

  v8f oacc[8];
  #pragma unroll
  for (int t = 0; t < 8; ++t) oacc[t] = (v8f){};
  float mrow[8], lrow[8];
  #pragma unroll
  for (int g = 0; g < 8; ++g) { mrow[g] = -1e30f; lrow[g] = 0.0f; }

  const int nkvb = (qblk + 1) * (BQ / BC);

  // ---- async stage: 2x B128 (K rows) + 2x B128 (V^T rows) per thread ----
  auto issue_stage = [&](int kvb, int buf) {
    const int kv0 = kvb * BC;
    bf16* kb = &k_s[buf][0];
    bf16* vb = &vt_s[buf][0];
    #pragma unroll
    for (int it = 0; it < 2; ++it) {           // K: 32 rows x 16 chunks of 16B
      const int i    = it * 256 + tid;
      const int row  = i >> 4;
      const int ch   = (i & 15) * 8;           // bf16 offset
      const bf16* g  = kp + (size_t)(kv0 + row) * D_ + ch;
      const unsigned l = (unsigned)(size_t)(kb + row * KS2 + ch);
      asm volatile("global_load_async_to_lds_b128 %0, %1, off"
                   :: "v"(l), "v"(g) : "memory");
    }
    #pragma unroll
    for (int it = 0; it < 2; ++it) {           // V^T: 128 rows x 4 chunks of 16B
      const int i    = it * 256 + tid;
      const int row  = i >> 2;
      const int ch   = (i & 3) * 8;            // bf16 offset within 32-wide row
      const bf16* g  = vp + (size_t)row * S_ + kv0 + ch;
      const unsigned l = (unsigned)(size_t)(vb + row * VS2 + ch);
      asm volatile("global_load_async_to_lds_b128 %0, %1, off"
                   :: "v"(l), "v"(g) : "memory");
    }
  };

  issue_stage(0, 0);

  for (int kvb = 0; kvb < nkvb; ++kvb) {
    const int kv0 = kvb * BC;
    const int cur = kvb & 1;
    const bool pre = (kvb + 1) < nkvb;
    if (pre) {
      issue_stage(kvb + 1, cur ^ 1);
      // 8 in flight; retire the 4 belonging to the current stage (in-order)
      asm volatile("s_wait_asynccnt 4" ::: "memory");
    } else {
      asm volatile("s_wait_asynccnt 0" ::: "memory");
    }
    __syncthreads();   // whole tile (written by all 8 waves) now visible

    if (kv0 <= qmax) {   // wave-uniform -> EXEC stays all-ones around WMMA
      const bf16* kbuf = &k_s[cur][0];
      const bf16* vbuf = &vt_s[cur][0];
      const int k0 = lhalf * 8;

      // ---- S = Q K^T : preload all 8 K-fragments, then drain 8 WMMAs ----
      v16bf kf[2][4];
      #pragma unroll
      for (int t = 0; t < 2; ++t) {
        const bf16* kr = kbuf + (t * 16 + l16) * KS2;
        #pragma unroll
        for (int c = 0; c < 4; ++c) {
          const int d0 = 32 * c + k0;
          kf[t][c] = cat8(*(const v8bf*)(kr + d0),
                          *(const v8bf*)(kr + d0 + 16));
        }
      }
      v8f sc[2];
      #pragma unroll
      for (int t = 0; t < 2; ++t) {
        v8f acc = (v8f){};
        #pragma unroll
        for (int c = 0; c < 4; ++c)
          acc = __builtin_amdgcn_wmma_f32_16x16x32_bf16(false, qf[c], false, kf[t][c],
                                                        (short)0, acc, false, false);
        sc[t] = acc;
      }

      // ---- causal mask (diagonal band only) ----
      if (kv0 + BC - 1 > qmin) {
        #pragma unroll
        for (int t = 0; t < 2; ++t) {
          const int kvpos = kv0 + t * 16 + l16;
          #pragma unroll
          for (int g = 0; g < 8; ++g) {
            const int qrow = qmin + g + lhalf * 8;
            if (kvpos > qrow) sc[t][g] = -1e30f;
          }
        }
      }

      // ---- online softmax (base-2), row reductions within 16-lane groups ----
      float pr[2][8];
      #pragma unroll
      for (int g = 0; g < 8; ++g) {
        float mx = fmaxf(sc[0][g], sc[1][g]);
        #pragma unroll
        for (int s = 8; s >= 1; s >>= 1) mx = fmaxf(mx, __shfl_xor(mx, s, 32));
        const float mnew = fmaxf(mrow[g], mx);
        const float corr = __builtin_amdgcn_exp2f(mrow[g] - mnew);
        mrow[g] = mnew;
        const float p0 = __builtin_amdgcn_exp2f(sc[0][g] - mnew);
        const float p1 = __builtin_amdgcn_exp2f(sc[1][g] - mnew);
        pr[0][g] = p0; pr[1][g] = p1;
        float rs = p0 + p1;
        #pragma unroll
        for (int s = 8; s >= 1; s >>= 1) rs += __shfl_xor(rs, s, 32);
        lrow[g] = lrow[g] * corr + rs;
        #pragma unroll
        for (int dt = 0; dt < 8; ++dt) oacc[dt][g] *= corr;
      }

      // ---- PV phase: batch V-fragment loads, then P via LDS, then 8 WMMAs ----
      v16bf vf[8];
      #pragma unroll
      for (int dt = 0; dt < 8; ++dt) {
        const bf16* vr = vbuf + (dt * 16 + l16) * VS2;
        vf[dt] = cat8(*(const v8bf*)(vr + k0),
                      *(const v8bf*)(vr + k0 + 16));
      }

      bf16* pw = &p_s[wid * 16 * VS2];
      #pragma unroll
      for (int t = 0; t < 2; ++t)
        #pragma unroll
        for (int g = 0; g < 8; ++g)
          pw[(g + lhalf * 8) * VS2 + t * 16 + l16] = (bf16)pr[t][g];

      v16bf pf = cat8(*(const v8bf*)(&pw[l16 * VS2 + k0]),
                      *(const v8bf*)(&pw[l16 * VS2 + k0 + 16]));

      #pragma unroll
      for (int dt = 0; dt < 8; ++dt)
        oacc[dt] = __builtin_amdgcn_wmma_f32_16x16x32_bf16(false, pf, false, vf[dt],
                                                           (short)0, oacc[dt], false, false);
    }
    __syncthreads();   // protect buffer reuse before next prefetch lands
  }

  // ---- normalize by 1/l and write O (fp32) ----
  #pragma unroll
  for (int g = 0; g < 8; ++g) {
    const float rl = 1.0f / lrow[g];
    float* orow = op + (size_t)(qmin + g + lhalf * 8) * D_;
    #pragma unroll
    for (int dt = 0; dt < 8; ++dt)
      orow[dt * 16 + l16] = oacc[dt][g] * rl;
  }
}

extern "C" void kernel_launch(void* const* d_in, const int* in_sizes, int n_in,
                              void* d_out, int out_size, void* d_ws, size_t ws_size,
                              hipStream_t stream) {
  (void)in_sizes; (void)n_in; (void)ws_size; (void)out_size;
  const float* q = (const float*)d_in[0];
  const float* k = (const float*)d_in[1];
  const float* v = (const float*)d_in[2];
  // d_in[3] is the causal mask; causality is computed analytically in-kernel.
  float* out = (float*)d_out;

  const size_t elems = (size_t)B_ * H_ * S_ * D_;     // 16.7M
  bf16* kws  = (bf16*)d_ws;                           // bf16 K, row-major
  bf16* vtws = kws + elems;                           // bf16 V^T, [bh][d][s]
  // requires ws_size >= 2 * elems * sizeof(bf16) = 64 MiB

  // pre-pass: K -> bf16 (elementwise), V -> bf16 transposed (LDS tile transpose)
  cvt_k_kernel<<<dim3((unsigned)(elems / 1024)), dim3(256), 0, stream>>>(k, kws);
  cvt_vt_kernel<<<dim3(B_ * H_ * (S_ / 32)), dim3(256), 0, stream>>>(v, vtws);

  // main attention
  fa_fwd_kernel<<<dim3(B_ * H_ * (S_ / BQ)), dim3(256), 0, stream>>>(q, kws, vtws, out);
}
